// SimVQ1D_11029476016540
// MI455X (gfx1250) — matrix operand order repaired and checked by reading, not compiled, not run on hardware
//
#include <hip/hip_runtime.h>
#include <hip/hip_bf16.h>
#include <float.h>
#include <stdint.h>

typedef __attribute__((ext_vector_type(16))) __bf16 v16bf;
typedef __attribute__((ext_vector_type(8)))  __bf16 v8bf;
typedef __attribute__((ext_vector_type(8)))  float  v8f;

#define ED   128     // embedding dim
#define NE   8192    // codebook entries
#define NZ   16384   // flattened z rows (8*2048)
#define TILE_N 64
#define LDS_STRIDE 272           // 256B row + 16B pad -> bank spread
#define TILE_BYTES (TILE_N * LDS_STRIDE)
#define NITER (NE / TILE_N)      // 128

__device__ __forceinline__ void async_wait0() {
#if __has_builtin(__builtin_amdgcn_s_wait_asynccnt)
  __builtin_amdgcn_s_wait_asynccnt(0);
#else
  asm volatile("s_wait_asynccnt 0x0" ::: "memory");
#endif
}

// ---------------------------------------------------------------------------
// Kernel 1: quant_codebook = emb @ W.T + b  (fp32), plus bf16 copy and ||c||^2
// ---------------------------------------------------------------------------
__global__ __launch_bounds__(256) void build_codebook_kernel(
    const float* __restrict__ emb, const float* __restrict__ W,
    const float* __restrict__ bias, float* __restrict__ Cf,
    __bf16* __restrict__ Cb, float* __restrict__ c2)
{
  __shared__ float Wl[ED * ED];          // 64 KB: full proj_w, row-major [j][k]
  const int t = threadIdx.x;
  for (int i = t; i < ED * ED / 4; i += 256)
    ((float4*)Wl)[i] = ((const float4*)W)[i];
  __syncthreads();

  const int r = blockIdx.x * 64 + (t >> 2);   // codebook row
  const int g = t & 3;                        // 32-col group
  const float* erow = emb + (size_t)r * ED;

  float acc[32];
  #pragma unroll
  for (int j = 0; j < 32; ++j) acc[j] = bias[g * 32 + j];

  for (int k0 = 0; k0 < ED; k0 += 32) {
    float e[32];
    #pragma unroll
    for (int i = 0; i < 8; ++i)
      ((float4*)e)[i] = ((const float4*)(erow + k0))[i];
    #pragma unroll
    for (int j = 0; j < 32; ++j) {
      const float4* wr = (const float4*)(Wl + (g * 32 + j) * ED + k0);
      float a = acc[j];
      #pragma unroll
      for (int k4 = 0; k4 < 8; ++k4) {
        float4 wv = wr[k4];
        a = fmaf(e[4 * k4 + 0], wv.x, a);
        a = fmaf(e[4 * k4 + 1], wv.y, a);
        a = fmaf(e[4 * k4 + 2], wv.z, a);
        a = fmaf(e[4 * k4 + 3], wv.w, a);
      }
      acc[j] = a;
    }
  }

  float s = 0.f;
  #pragma unroll
  for (int j = 0; j < 32; ++j) {
    float v = acc[j];
    Cf[(size_t)r * ED + g * 32 + j] = v;
    Cb[(size_t)r * ED + g * 32 + j] = (__bf16)v;
    s = fmaf(v, v, s);
  }
  // 4 consecutive lanes hold partial sums for the same row
  s += __shfl_xor(s, 1, 32);
  s += __shfl_xor(s, 2, 32);
  if (g == 0) c2[r] = s;
}

// ---------------------------------------------------------------------------
// Kernel 2: fused distance GEMM (bf16 WMMA, f32 acc) + argmin + gather.
// Block = 4 waves; each wave owns TWO 16-row z tiles (32 rows) so every B
// fragment feeds two independent WMMA accumulator chains. Codebook streamed
// through a double-buffered LDS tile via global_load_async_to_lds_b128
// (ASYNCcnt path); one barrier per 64-row tile.
// ---------------------------------------------------------------------------
__global__ __launch_bounds__(128) void vq_argmin_kernel(
    const float* __restrict__ z, const float* __restrict__ Cf,
    const __bf16* __restrict__ Cb, const float* __restrict__ c2,
    float* __restrict__ zq, float* __restrict__ idx_out)
{
  __shared__ __align__(16) char ldsB[2 * TILE_BYTES];

  const int t  = threadIdx.x;
  const int w  = t >> 5;          // wave in block (0..3)
  const int l  = t & 31;          // lane
  const int lo = l & 15;          // M-row (A) / N-col (B,C,D) within tile
  const int hi = l >> 4;          // half selector
  const int m0 = blockIdx.x * 128 + w * 32;   // first of 32 rows for this wave

  // Per-thread async-copy slots: 64 rows x 256B = 1024 x 16B chunks, 8/thread.
  const int row0 = t >> 4;        // rows row0 + c*8, c = 0..7
  const int col0 = t & 15;
  const uint32_t lds_base =
      (uint32_t)(uintptr_t)(ldsB) + (uint32_t)(row0 * LDS_STRIDE + col0 * 16);

  // --- A fragments for both M-tiles: z rows -> bf16, ISA 16-bit A layout:
  // lane<16: K = kb..kb+7, kb+16..kb+23 with kb=0; lane>=16: kb=8.
  const int kb = hi * 8;
  v16bf afr0[4], afr1[4];
  #pragma unroll
  for (int mt = 0; mt < 2; ++mt) {
    const float* zrow = z + (size_t)(m0 + mt * 16 + lo) * ED;
    #pragma unroll
    for (int kk = 0; kk < 4; ++kk) {
      float f[16];
      ((float4*)f)[0] = *(const float4*)(zrow + kk * 32 + kb);
      ((float4*)f)[1] = *(const float4*)(zrow + kk * 32 + kb + 4);
      ((float4*)f)[2] = *(const float4*)(zrow + kk * 32 + kb + 16);
      ((float4*)f)[3] = *(const float4*)(zrow + kk * 32 + kb + 20);
      v16bf a;
      #pragma unroll
      for (int i = 0; i < 16; ++i) a[i] = (__bf16)f[i];
      (mt ? afr1 : afr0)[kk] = a;
    }
  }

  float minval[2][8];
  int   minidx[2][8];
  #pragma unroll
  for (int mt = 0; mt < 2; ++mt)
    #pragma unroll
    for (int v = 0; v < 8; ++v) { minval[mt][v] = FLT_MAX; minidx[mt][v] = 0; }

  // Issue async copy of one 64-row bf16 codebook tile into LDS buffer `buf`.
  auto issue_tile = [&](int it, int buf) {
    #pragma unroll
    for (int c = 0; c < 8; ++c) {
      const int row = row0 + c * 8;
      const uint64_t g = (uint64_t)(uintptr_t)
          (Cb + (size_t)(it * TILE_N + row) * ED + col0 * 8);
      const uint32_t lds = lds_base + (uint32_t)(buf * TILE_BYTES + c * 8 * LDS_STRIDE);
      asm volatile("global_load_async_to_lds_b128 %0, %1, off"
                   :: "v"(lds), "v"(g) : "memory");
    }
  };

  // Prologue: bring in tile 0.
  issue_tile(0, 0);
  async_wait0();
  __syncthreads();

  for (int it = 0; it < NITER; ++it) {
    const int cur = it & 1;
    if (it + 1 < NITER) issue_tile(it + 1, cur ^ 1);   // overlap with compute

    const char* tile = ldsB + cur * TILE_BYTES;
    #pragma unroll
    for (int sub = 0; sub < 4; ++sub) {
      const int nb = it * TILE_N + sub * 16;
      const float c2v = c2[nb + lo];
      // Preload all 4 B fragments (lane = N codebook row, half picks K 8h..8h+7, 16+8h..16+8h+7)
      const char* bb = tile + (sub * 16 + lo) * LDS_STRIDE + hi * 16;
      v16bf bfr[4];
      #pragma unroll
      for (int kk = 0; kk < 4; ++kk) {
        v8bf b0 = *(const v8bf*)(bb + kk * 64);
        v8bf b1 = *(const v8bf*)(bb + kk * 64 + 32);
        bfr[kk] = __builtin_shufflevector(b0, b1,
            0, 1, 2, 3, 4, 5, 6, 7, 8, 9, 10, 11, 12, 13, 14, 15);
      }
      // Two independent accumulator chains, each B fragment used twice.
      v8f acc0 = {}, acc1 = {};
      #pragma unroll
      for (int kk = 0; kk < 4; ++kk) {
        acc0 = __builtin_amdgcn_wmma_f32_16x16x32_bf16(
            false, afr0[kk], false, bfr[kk], (short)0, acc0, false, false);
        acc1 = __builtin_amdgcn_wmma_f32_16x16x32_bf16(
            false, afr1[kk], false, bfr[kk], (short)0, acc1, false, false);
      }
      // dist = ||c||^2 - 2*z.c  (||z||^2 dropped: constant per row)
      #pragma unroll
      for (int v = 0; v < 8; ++v) {
        float d0 = fmaf(-2.0f, acc0[v], c2v);
        float d1 = fmaf(-2.0f, acc1[v], c2v);
        if (d0 < minval[0][v]) { minval[0][v] = d0; minidx[0][v] = nb + lo; }
        if (d1 < minval[1][v]) { minval[1][v] = d1; minidx[1][v] = nb + lo; }
      }
    }

    async_wait0();      // next tile's LDS writes have landed
    __syncthreads();    // all waves done reading `cur`; writes visible
  }

  // --- reduce argmin across the 16 lanes of each half (rows v and v+8),
  //     write indices, and gather fp32 codebook rows as z_q.
  #pragma unroll
  for (int mt = 0; mt < 2; ++mt) {
    const int mb = m0 + mt * 16;
    #pragma unroll
    for (int v = 0; v < 8; ++v) {
      float mv = minval[mt][v];
      int   mi = minidx[mt][v];
      #pragma unroll
      for (int mk = 1; mk < 16; mk <<= 1) {
        float ov = __shfl_xor(mv, mk, 32);
        int   oi = __shfl_xor(mi, mk, 32);
        if (ov < mv || (ov == mv && oi < mi)) { mv = ov; mi = oi; }
      }
      const int i0 = __shfl(mi, 0, 32);    // winner for row mb+v
      const int i1 = __shfl(mi, 16, 32);   // winner for row mb+8+v
      if (l == 0)  idx_out[mb + v]     = (float)mi;
      if (l == 16) idx_out[mb + 8 + v] = (float)mi;
      // each lane copies one float4 -> 32 lanes cover the 128-float row
      ((float4*)(zq + (size_t)(mb + v) * ED))[l] =
          ((const float4*)(Cf + (size_t)i0 * ED))[l];
      ((float4*)(zq + (size_t)(mb + 8 + v) * ED))[l] =
          ((const float4*)(Cf + (size_t)i1 * ED))[l];
    }
  }
}

// ---------------------------------------------------------------------------
extern "C" void kernel_launch(void* const* d_in, const int* in_sizes, int n_in,
                              void* d_out, int out_size, void* d_ws, size_t ws_size,
                              hipStream_t stream) {
  const float* zp   = (const float*)d_in[0];   // [8,2048,128]
  const float* emb  = (const float*)d_in[1];   // [8192,128]
  const float* W    = (const float*)d_in[2];   // [128,128]
  const float* bias = (const float*)d_in[3];   // [128]

  float* out    = (float*)d_out;
  float* zq     = out;                         // 16384*128 floats
  float* idxout = out + (size_t)NZ * ED;       // 16384 indices (as float)

  char* ws = (char*)d_ws;                      // 6.03 MB used
  float*  Cf = (float*)ws;                                   // 4 MB fp32 codebook
  __bf16* Cb = (__bf16*)(ws + (size_t)NE * ED * 4);          // 2 MB bf16 codebook
  float*  c2 = (float*)(ws + (size_t)NE * ED * 6);           // 32 KB ||c||^2

  build_codebook_kernel<<<NE / 64, 256, 0, stream>>>(emb, W, bias, Cf, Cb, c2);
  vq_argmin_kernel<<<NZ / 128, 128, 0, stream>>>(zp, Cf, Cb, c2, zq, idxout);
}
